// TransformerBlock_65481071397684
// MI455X (gfx1250) — matrix operand order repaired and checked by reading, not compiled, not run on hardware
//
#include <hip/hip_runtime.h>
#include <hip/hip_bf16.h>

// ---------------- Problem constants (from reference) ----------------
#define E_DIM   1024
#define H_DIM   16
#define DK_DIM  64
#define FFN_DIM 4096
#define B_DIM   2
#define T_DIM   2048
#define N_ROWS  (B_DIM * T_DIM)   // 4096

// ---------------- WMMA types (gfx1250, wave32) ----------------
typedef __attribute__((ext_vector_type(16))) __bf16 v16bf;
typedef __attribute__((ext_vector_type(8)))  float  v8f;

struct Frag32B { uint4 lo, hi; };

static __device__ __forceinline__ unsigned short f2bf(float f) {
    unsigned int u = __builtin_bit_cast(unsigned int, f);
    u += 0x7FFFu + ((u >> 16) & 1u);           // round-to-nearest-even
    return (unsigned short)(u >> 16);
}

static __device__ __forceinline__ v16bf frag_from(uint4 lo, uint4 hi) {
    Frag32B t{lo, hi};
    return __builtin_bit_cast(v16bf, t);
}

static __device__ __forceinline__ v8f wmma_bf16(v16bf a, v16bf b, v8f c) {
    return __builtin_amdgcn_wmma_f32_16x16x32_bf16(false, a, false, b, (short)0, c,
                                                   false, false);
}

// ---------------- fp32 -> bf16 conversion kernels ----------------
__global__ void f2bf_copy(const float* __restrict__ src,
                          unsigned short* __restrict__ dst, size_t n) {
    size_t i = (size_t)blockIdx.x * blockDim.x + threadIdx.x;
    size_t stride = (size_t)gridDim.x * blockDim.x;
    for (; i < n; i += stride) dst[i] = f2bf(src[i]);
}

// dst[c*R + r] = bf16(src[r*C + c])   (src is [R,C])
__global__ void f2bf_transpose(const float* __restrict__ src,
                               unsigned short* __restrict__ dst, int R, int C) {
    size_t n = (size_t)R * C;
    size_t i = (size_t)blockIdx.x * blockDim.x + threadIdx.x;
    size_t stride = (size_t)gridDim.x * blockDim.x;
    for (; i < n; i += stride) {
        int r = (int)(i / C), c = (int)(i % C);
        dst[(size_t)c * R + r] = f2bf(src[i]);
    }
}

// ---------------- Generic bf16 WMMA GEMM ----------------
// out[m,n] = sum_k A[m,k] * Bc[n,k]  (Bc row n == B-matrix column n, contiguous)
// Wave tile 32x64 (8 accumulators), 8 waves as 4x2 -> 128x128 block tile.
// modes: 0 = bf16 out (+bias), 1 = bf16 out ReLU(+bias), 2 = f32 out (+bias),
//        3 = bf16 scatter to [B,H,T,DK] (Q/K), 4 = bf16 scatter to [B,H,DK,T] (V^T)
__global__ __launch_bounds__(256) void gemm_bf16_wmma(
    const unsigned short* __restrict__ A,
    const unsigned short* __restrict__ Bc,
    const float* __restrict__ bias,
    void* __restrict__ out,
    int M, int N, int K, int mode)
{
    const int lane = threadIdx.x & 31;
    const int w    = threadIdx.x >> 5;
    const int wr   = w >> 1, wc = w & 1;          // 4x2 wave grid -> 128x128 block tile
    const int m0   = blockIdx.x * 128 + wr * 32;
    const int n0   = blockIdx.y * 128 + wc * 64;
    const int l16  = lane & 15;
    const int hi16 = (lane >> 4) & 1;

    v8f c[8];
    for (int i = 0; i < 8; ++i) c[i] = {};

    const size_t arow0 = (size_t)(m0 + l16) * K;
    const size_t arow1 = (size_t)(m0 + 16 + l16) * K;
    size_t brow[4];
    for (int j = 0; j < 4; ++j) brow[j] = (size_t)(n0 + j * 16 + l16) * K;
    const int aoff = hi16 ? 16 : 0;   // A lane-half K split: K 0..15 vs 16..31
    const int boff = hi16 ? 16 : 0;   // B lane-half K split: K 0..15 vs 16..31

    for (int kb = 0; kb < K; kb += 32) {
        const uint4* pa0 = (const uint4*)(A + arow0 + kb + aoff);
        const uint4* pa1 = (const uint4*)(A + arow1 + kb + aoff);
        v16bf a0 = frag_from(pa0[0], pa0[1]);   // 16 contiguous K values
        v16bf a1 = frag_from(pa1[0], pa1[1]);
        v16bf b[4];
        for (int j = 0; j < 4; ++j) {
            const uint4* pb = (const uint4*)(Bc + brow[j] + kb + boff);
            b[j] = frag_from(pb[0], pb[1]);     // 16 contiguous K values
        }
        // prefetch the streamed rows ~8 iterations ahead (speculative, OOB-safe)
        __builtin_prefetch(A + arow0 + kb + 256, 0, 1);
        __builtin_prefetch(A + arow1 + kb + 256, 0, 1);
        __builtin_prefetch(Bc + brow[0] + kb + 256, 0, 1);
        __builtin_prefetch(Bc + brow[2] + kb + 256, 0, 1);
        c[0] = wmma_bf16(a0, b[0], c[0]);
        c[1] = wmma_bf16(a0, b[1], c[1]);
        c[2] = wmma_bf16(a0, b[2], c[2]);
        c[3] = wmma_bf16(a0, b[3], c[3]);
        c[4] = wmma_bf16(a1, b[0], c[4]);
        c[5] = wmma_bf16(a1, b[1], c[5]);
        c[6] = wmma_bf16(a1, b[2], c[6]);
        c[7] = wmma_bf16(a1, b[3], c[7]);
    }

    float bv[4] = {0.f, 0.f, 0.f, 0.f};
    if (bias)
        for (int j = 0; j < 4; ++j) bv[j] = bias[n0 + j * 16 + l16];

    for (int j = 0; j < 8; ++j) {
        const int jr = j >> 2, jc = j & 3;
        const int col = n0 + jc * 16 + l16;
        const float badd = bv[jc];
        for (int r = 0; r < 8; ++r) {
            const int row = m0 + jr * 16 + r + hi16 * 8;   // C layout: VGPR r -> M=r / r+8
            float v = c[j][r] + badd;
            if (mode == 0) {
                ((unsigned short*)out)[(size_t)row * N + col] = f2bf(v);
            } else if (mode == 1) {
                v = fmaxf(v, 0.0f);
                ((unsigned short*)out)[(size_t)row * N + col] = f2bf(v);
            } else if (mode == 2) {
                ((float*)out)[(size_t)row * N + col] = v;
            } else {
                const int b2 = row >> 11, t = row & (T_DIM - 1);
                const int h = col >> 6,  dk = col & (DK_DIM - 1);
                size_t idx;
                if (mode == 3)       // [B,H,T,DK]
                    idx = (((size_t)(b2 * H_DIM + h) * T_DIM) + t) * DK_DIM + dk;
                else                 // [B,H,DK,T]
                    idx = (((size_t)(b2 * H_DIM + h) * DK_DIM) + dk) * T_DIM + t;
                ((unsigned short*)out)[idx] = f2bf(v);
            }
        }
    }
}

// ---------------- Flash attention (bf16 WMMA, online softmax) ----------------
// q,k: [B,H,T,DK] bf16 ; vT: [B,H,DK,T] bf16 ; out attnb: [N,E] bf16
__global__ __launch_bounds__(256) void attention_wmma(
    const unsigned short* __restrict__ q,
    const unsigned short* __restrict__ k,
    const unsigned short* __restrict__ vT,
    const float* __restrict__ gate,
    unsigned short* __restrict__ attnb)
{
    __shared__ unsigned short plds[8 * 512];   // per-wave 16x32 bf16 P tile
    const int lane = threadIdx.x & 31;
    const int w    = threadIdx.x >> 5;
    const int l16  = lane & 15;
    const int hi16 = (lane >> 4) & 1;
    const int bh   = blockIdx.y;
    const int b    = bh >> 4, h = bh & 15;
    const int q0   = blockIdx.x * 128 + w * 16;       // 16 query rows per wave
    const float sg = 1.0f / (1.0f + __expf(-gate[h]));

    const unsigned short* qh = q  + (size_t)bh * T_DIM * DK_DIM;
    const unsigned short* kh = k  + (size_t)bh * T_DIM * DK_DIM;
    const unsigned short* vh = vT + (size_t)bh * DK_DIM * T_DIM;
    unsigned short* pl = plds + w * 512;

    // Q A-fragments for K-chunks 0..31 and 32..63 of DK
    const uint4* pq0 = (const uint4*)(qh + (size_t)(q0 + l16) * DK_DIM + (hi16 ? 16 : 0));
    const uint4* pq1 = (const uint4*)(qh + (size_t)(q0 + l16) * DK_DIM + 32 + (hi16 ? 16 : 0));
    const v16bf qa0 = frag_from(pq0[0], pq0[1]);
    const v16bf qa1 = frag_from(pq1[0], pq1[1]);

    float mrow[8], lrow[8], alpha[8], pv0[8], pv1[8];
    v8f o0 = {}, o1 = {}, o2 = {}, o3 = {};
    for (int r = 0; r < 8; ++r) { mrow[r] = -1e30f; lrow[r] = 0.f; }

    for (int t0 = 0; t0 < T_DIM; t0 += 32) {
        // ---- scores: S[16x32] = Q[16x64] . K_tile[32x64]^T (B columns = K rows)
        v8f s0 = {}, s1 = {};
        {
            const uint4* pb = (const uint4*)(kh + (size_t)(t0 + l16) * DK_DIM + (hi16 ? 16 : 0));
            s0 = wmma_bf16(qa0, frag_from(pb[0], pb[1]), s0);
            const uint4* pc = (const uint4*)(kh + (size_t)(t0 + l16) * DK_DIM + 32 + (hi16 ? 16 : 0));
            s0 = wmma_bf16(qa1, frag_from(pc[0], pc[1]), s0);
        }
        {
            const uint4* pb = (const uint4*)(kh + (size_t)(t0 + 16 + l16) * DK_DIM + (hi16 ? 16 : 0));
            s1 = wmma_bf16(qa0, frag_from(pb[0], pb[1]), s1);
            const uint4* pc = (const uint4*)(kh + (size_t)(t0 + 16 + l16) * DK_DIM + 32 + (hi16 ? 16 : 0));
            s1 = wmma_bf16(qa1, frag_from(pc[0], pc[1]), s1);
        }
        // prefetch next key/value tiles (speculative)
        __builtin_prefetch(kh + (size_t)(t0 + 32 + l16) * DK_DIM, 0, 1);
        __builtin_prefetch(vh + (size_t)(l16)*T_DIM + t0 + 32, 0, 1);

        // ---- online softmax (row r lives in lanes of this 16-lane half)
        const float scale = 0.125f;  // 1/sqrt(64)
        for (int r = 0; r < 8; ++r) {
            float x0 = s0[r] * scale, x1 = s1[r] * scale;
            float t = fmaxf(x0, x1);
            for (int off = 8; off; off >>= 1) t = fmaxf(t, __shfl_xor(t, off, 32));
            float mn = fmaxf(mrow[r], t);
            float a  = __expf(mrow[r] - mn);
            float p0 = __expf(x0 - mn);
            float p1 = __expf(x1 - mn);
            float rs = p0 + p1;
            for (int off = 8; off; off >>= 1) rs += __shfl_xor(rs, off, 32);
            lrow[r] = lrow[r] * a + rs;
            mrow[r] = mn;
            alpha[r] = a; pv0[r] = p0; pv1[r] = p1;
        }
        for (int r = 0; r < 8; ++r) {
            o0[r] *= alpha[r]; o1[r] *= alpha[r];
            o2[r] *= alpha[r]; o3[r] *= alpha[r];
        }

        // ---- C-layout P -> LDS row-major -> A-layout fragment
        for (int r = 0; r < 8; ++r) {
            const int row = r + hi16 * 8;
            pl[row * 32 + l16]      = f2bf(pv0[r]);
            pl[row * 32 + 16 + l16] = f2bf(pv1[r]);
        }
        __syncthreads();
        const uint4* pp = (const uint4*)(pl + l16 * 32 + (hi16 ? 16 : 0));
        const v16bf pa = frag_from(pp[0], pp[1]);
        __syncthreads();

        // ---- out += P[16x32] . V_tile[32x64]; B columns = vT rows (contiguous in t)
        {
            const uint4* pv = (const uint4*)(vh + (size_t)(0 * 16 + l16) * T_DIM + t0 + (hi16 ? 16 : 0));
            o0 = wmma_bf16(pa, frag_from(pv[0], pv[1]), o0);
        }
        {
            const uint4* pv = (const uint4*)(vh + (size_t)(1 * 16 + l16) * T_DIM + t0 + (hi16 ? 16 : 0));
            o1 = wmma_bf16(pa, frag_from(pv[0], pv[1]), o1);
        }
        {
            const uint4* pv = (const uint4*)(vh + (size_t)(2 * 16 + l16) * T_DIM + t0 + (hi16 ? 16 : 0));
            o2 = wmma_bf16(pa, frag_from(pv[0], pv[1]), o2);
        }
        {
            const uint4* pv = (const uint4*)(vh + (size_t)(3 * 16 + l16) * T_DIM + t0 + (hi16 ? 16 : 0));
            o3 = wmma_bf16(pa, frag_from(pv[0], pv[1]), o3);
        }
    }

    // ---- normalize, gate, write [N,E] bf16
    for (int r = 0; r < 8; ++r) {
        const float inv = (lrow[r] > 0.f) ? (sg / lrow[r]) : 0.f;
        const int trow = q0 + r + hi16 * 8;
        const size_t orow = ((size_t)b * T_DIM + trow) * E_DIM + (size_t)h * DK_DIM;
        attnb[orow +  0 + l16] = f2bf(o0[r] * inv);
        attnb[orow + 16 + l16] = f2bf(o1[r] * inv);
        attnb[orow + 32 + l16] = f2bf(o2[r] * inv);
        attnb[orow + 48 + l16] = f2bf(o3[r] * inv);
    }
}

// ---------------- Residual + LayerNorm (one block per row) ----------------
__global__ __launch_bounds__(256) void ln_residual(
    const float* __restrict__ xa, const float* __restrict__ xb,
    const float* __restrict__ g,  const float* __restrict__ bb,
    float* __restrict__ of, unsigned short* __restrict__ ob)
{
    __shared__ float s1[256], s2[256];
    const int row = blockIdx.x;
    const int tid = threadIdx.x;
    const size_t base = (size_t)row * E_DIM;
    float v[4];
    float a1 = 0.f, a2 = 0.f;
    for (int i = 0; i < 4; ++i) {
        const int c = tid + i * 256;
        v[i] = xa[base + c] + xb[base + c];
        a1 += v[i]; a2 += v[i] * v[i];
    }
    s1[tid] = a1; s2[tid] = a2;
    __syncthreads();
    for (int off = 128; off; off >>= 1) {
        if (tid < off) { s1[tid] += s1[tid + off]; s2[tid] += s2[tid + off]; }
        __syncthreads();
    }
    const float mu  = s1[0] * (1.0f / E_DIM);
    const float var = s2[0] * (1.0f / E_DIM) - mu * mu;
    const float rs  = rsqrtf(var + 1e-5f);
    for (int i = 0; i < 4; ++i) {
        const int c = tid + i * 256;
        const float o = (v[i] - mu) * rs * g[c] + bb[c];
        of[base + c] = o;
        if (ob) ob[base + c] = f2bf(o);
    }
}

// ---------------- Host-side orchestration ----------------
extern "C" void kernel_launch(void* const* d_in, const int* in_sizes, int n_in,
                              void* d_out, int out_size, void* d_ws, size_t ws_size,
                              hipStream_t stream)
{
    const float* x    = (const float*)d_in[0];
    const float* Wq   = (const float*)d_in[1];
    const float* Wk   = (const float*)d_in[2];
    const float* Wv   = (const float*)d_in[3];
    const float* Wo   = (const float*)d_in[4];
    const float* bo   = (const float*)d_in[5];
    const float* gate = (const float*)d_in[6];
    const float* g1   = (const float*)d_in[7];
    const float* b1n  = (const float*)d_in[8];
    const float* W1   = (const float*)d_in[9];
    const float* b1   = (const float*)d_in[10];
    const float* W2   = (const float*)d_in[11];
    const float* b2   = (const float*)d_in[12];
    const float* g2   = (const float*)d_in[13];
    const float* b2n  = (const float*)d_in[14];

    char* p = (char*)d_ws;
    auto alloc = [&](size_t bytes) -> void* {
        void* r = (void*)p;
        p += (bytes + 255) & ~(size_t)255;
        return r;
    };
    unsigned short* xb    = (unsigned short*)alloc((size_t)N_ROWS * E_DIM * 2);
    unsigned short* Wqb   = (unsigned short*)alloc((size_t)E_DIM * E_DIM * 2);
    unsigned short* Wkb   = (unsigned short*)alloc((size_t)E_DIM * E_DIM * 2);
    unsigned short* Wvb   = (unsigned short*)alloc((size_t)E_DIM * E_DIM * 2);
    unsigned short* Wob   = (unsigned short*)alloc((size_t)E_DIM * E_DIM * 2);
    unsigned short* W1t   = (unsigned short*)alloc((size_t)E_DIM * FFN_DIM * 2);
    unsigned short* W2t   = (unsigned short*)alloc((size_t)FFN_DIM * E_DIM * 2);
    unsigned short* qbuf  = (unsigned short*)alloc((size_t)N_ROWS * E_DIM * 2);
    unsigned short* kbuf  = (unsigned short*)alloc((size_t)N_ROWS * E_DIM * 2);
    unsigned short* vtbuf = (unsigned short*)alloc((size_t)N_ROWS * E_DIM * 2);
    unsigned short* attnb = (unsigned short*)alloc((size_t)N_ROWS * E_DIM * 2);
    float*          aout  = (float*)alloc((size_t)N_ROWS * E_DIM * 4);
    float*          hf    = (float*)alloc((size_t)N_ROWS * E_DIM * 4);
    unsigned short* hb    = (unsigned short*)alloc((size_t)N_ROWS * E_DIM * 2);
    unsigned short* ffn1  = (unsigned short*)alloc((size_t)N_ROWS * FFN_DIM * 2);
    float*          fout  = (float*)alloc((size_t)N_ROWS * E_DIM * 4);

    const dim3 blk(256);
    const int cvt = 2048;

    // Precision conversion (bf16 matmul path; fp32 accumulate everywhere)
    f2bf_copy<<<cvt, blk, 0, stream>>>(x,  xb,  (size_t)N_ROWS * E_DIM);
    f2bf_copy<<<cvt, blk, 0, stream>>>(Wq, Wqb, (size_t)E_DIM * E_DIM);
    f2bf_copy<<<cvt, blk, 0, stream>>>(Wk, Wkb, (size_t)E_DIM * E_DIM);
    f2bf_copy<<<cvt, blk, 0, stream>>>(Wv, Wvb, (size_t)E_DIM * E_DIM);
    f2bf_copy<<<cvt, blk, 0, stream>>>(Wo, Wob, (size_t)E_DIM * E_DIM);
    f2bf_transpose<<<cvt, blk, 0, stream>>>(W1, W1t, E_DIM, FFN_DIM);   // [FFN,E]
    f2bf_transpose<<<cvt, blk, 0, stream>>>(W2, W2t, FFN_DIM, E_DIM);   // [E,FFN]

    // QKV projections (x @ W.T): scatter Q,K -> [B,H,T,DK]; V -> [B,H,DK,T]
    gemm_bf16_wmma<<<dim3(N_ROWS / 128, E_DIM / 128), blk, 0, stream>>>(
        xb, Wqb, nullptr, qbuf, N_ROWS, E_DIM, E_DIM, 3);
    gemm_bf16_wmma<<<dim3(N_ROWS / 128, E_DIM / 128), blk, 0, stream>>>(
        xb, Wkb, nullptr, kbuf, N_ROWS, E_DIM, E_DIM, 3);
    gemm_bf16_wmma<<<dim3(N_ROWS / 128, E_DIM / 128), blk, 0, stream>>>(
        xb, Wvb, nullptr, vtbuf, N_ROWS, E_DIM, E_DIM, 4);

    // Attention with head-gating -> [N,E] bf16
    attention_wmma<<<dim3(T_DIM / 128, B_DIM * H_DIM), blk, 0, stream>>>(
        qbuf, kbuf, vtbuf, gate, attnb);

    // Output projection (+bo) -> fp32
    gemm_bf16_wmma<<<dim3(N_ROWS / 128, E_DIM / 128), blk, 0, stream>>>(
        attnb, Wob, bo, aout, N_ROWS, E_DIM, E_DIM, 2);

    // h = LN(x + attn_out) -> fp32 hf + bf16 hb
    ln_residual<<<N_ROWS, blk, 0, stream>>>(x, aout, g1, b1n, hf, hb);

    // ffn1 = relu(h @ W1 + b1) -> bf16 [N,FFN]
    gemm_bf16_wmma<<<dim3(N_ROWS / 128, FFN_DIM / 128), blk, 0, stream>>>(
        hb, W1t, b1, ffn1, N_ROWS, FFN_DIM, E_DIM, 1);

    // ffn = ffn1 @ W2 + b2 -> fp32 [N,E]
    gemm_bf16_wmma<<<dim3(N_ROWS / 128, E_DIM / 128), blk, 0, stream>>>(
        ffn1, W2t, b2, fout, N_ROWS, E_DIM, FFN_DIM, 2);

    // out = LN(h + ffn) -> d_out fp32
    ln_residual<<<N_ROWS, blk, 0, stream>>>(hf, fout, g2, b2n, (float*)d_out, nullptr);
}
